// QuantizedHolographicTTEmbedding_41575283425407
// MI455X (gfx1250) — matrix operand order repaired and checked by reading, not compiled, not run on hardware
//
#include <hip/hip_runtime.h>
#include <hip/hip_bf16.h>
#include <stdint.h>

// ---------------------------------------------------------------------------
// QuantizedHolographicTTEmbedding for MI455X (gfx1250, wave32, WMMA)
//
// Kernel 1: dequantize both TT cores ONCE (449 rows, fold scale/zp/cos(phase))
//           into d_ws, pre-swizzled into v_wmma_f32_16x16x32_f16 fragment
//           order so the hot loop does zero layout work.
// Kernel 2: one wave per token: 16 lane-contiguous global_load_b128 fragment
//           loads (L2-resident), 8 WMMAs, LDS-staged coalesced NT stores of
//           the 64 MB fp32 output (the true bandwidth bound, ~3 us floor).
// ---------------------------------------------------------------------------

typedef __attribute__((ext_vector_type(16))) _Float16 v16h;
typedef __attribute__((ext_vector_type(8)))  _Float16 v8h;
typedef __attribute__((ext_vector_type(8)))  float    v8f;
typedef __attribute__((ext_vector_type(4)))  float    v4f;

namespace {
constexpr int V1   = 225;
constexpr int V2   = 224;
constexpr int RANK = 64;
constexpr int DM   = 1024;       // D1*D2 == d_model
constexpr int TOKENS = 4 * 4096; // B * L
constexpr int WAVES_PER_BLOCK = 8;
constexpr int ROW_HALFS = 2048;  // 4 frags * 32 lanes * 16 halfs per core row
}

// ---------------------------------------------------------------------------
// Fragment layouts (wave32):
//   A (16-bit, 16x32, MxK): lane(lo=lane&15, hi=lane>>4) holds M=lo;
//     half j<8  -> K = hi*8 + j          (VGPRs 0-3)
//     half j>=8 -> K = 16 + hi*8 + (j-8) (VGPRs 4-7)
//   B (16-bit, 32x16, KxN): lane holds N=lo; half j -> same K as A
// ws layout per core row (2048 halfs = 4 KB):
//   byte offset = frag*1024 + chunk*512 + lane*16 + j8*2,  j = chunk*8 + j8
//   frag = t{m,n}*2 + ks   (ks = K/32 step)
// => each 16-half fragment is two lane-contiguous 512 B global_load_b128's.
// ---------------------------------------------------------------------------

__global__ __launch_bounds__(256)
void tt_dequant_swizzle_kernel(const signed char* __restrict__ c1q,  // [V1][64][32]
                               const float*       __restrict__ s1p,
                               const float*       __restrict__ z1p,
                               const signed char* __restrict__ c2q,  // [V2][64][32]
                               const float*       __restrict__ s2p,
                               const float*       __restrict__ z2p,
                               const float*       __restrict__ phase, // [64]
                               _Float16*          __restrict__ ws)
{
    const int row  = blockIdx.x;          // [0, V1+V2)
    const bool isA = row < V1;
    const int idx  = isA ? row : row - V1;

    const signed char* src = (isA ? c1q : c2q) + (size_t)idx * (RANK * 32);
    _Float16*          dst = ws + (size_t)row * ROW_HALFS; // A rows then B rows

    const float scale = isA ? *s1p : *s2p;
    const float zp    = isA ? *z1p : *z2p;

    // Each thread produces 8 consecutive halfs = one 16 B store.
    const int o0   = threadIdx.x * 8;     // [0, 2048)
    const int f    = o0 >> 9;             // frag index 0..3
    const int rem  = o0 & 511;
    const int ch   = rem >> 8;            // chunk 0/1
    const int lane = (rem & 255) >> 3;    // target lane 0..31
    const int lo   = lane & 15;
    const int hi   = lane >> 4;
    const int tmn  = f >> 1;              // tile along M (A) or N (B)
    const int ks   = f & 1;               // K step (0: K 0..31, 1: K 32..63)
    const int mn   = tmn * 16 + lo;       // M (A) or N (B)

    v8h outv;
    #pragma unroll
    for (int j8 = 0; j8 < 8; ++j8) {
        const int j = ch * 8 + j8;        // half index within fragment, 0..15
        int k;
        if (isA) k = ks * 32 + ((j < 8) ? (hi * 8 + j) : (16 + hi * 8 + (j - 8)));
        else     k = ks * 32 + ((j < 8) ? (hi * 8 + j) : (16 + hi * 8 + (j - 8)));

        float v = ((float)src[k * 32 + mn] - zp) * scale;
        if (isA) v *= __cosf(phase[k]);   // fold phase modulation into core1
        outv[j8] = (_Float16)v;
    }
    *(v8h*)((char*)dst + f * 1024 + ch * 512 + lane * 16) = outv;
}

__global__ __launch_bounds__(256)
void tt_embed_wmma_kernel(const int*      __restrict__ ids,
                          const _Float16* __restrict__ ws,   // swizzled f16 cores
                          float*          __restrict__ out)  // [TOKENS][1024]
{
    __shared__ __align__(16) float o_t[WAVES_PER_BLOCK][DM];

    const int lane = threadIdx.x & 31;   // wave32
    const int wave = threadIdx.x >> 5;
    const int tok  = blockIdx.x * WAVES_PER_BLOCK + wave;

    const int id = ids[tok];
    int i1 = id / V2; if (i1 > V1 - 1) i1 = V1 - 1;
    const int i2 = id % V2;

    // v8h-unit pointers into the swizzled cores (L2-resident)
    const v8h* aB = (const v8h*)ws + (size_t)i1 * 256;               // 2048 halfs
    const v8h* bB = (const v8h*)ws + ((size_t)V1 + i2) * 256;

    // Load all 8 fragments: 16 lane-contiguous global_load_b128 (512 B each)
    v16h af[2][2], bf[2][2];
    #pragma unroll
    for (int t = 0; t < 2; ++t)
        #pragma unroll
        for (int ks = 0; ks < 2; ++ks) {
            const int f = t * 2 + ks;
            const v8h a_lo = aB[f * 64 + lane];
            const v8h a_hi = aB[f * 64 + 32 + lane];
            af[t][ks] = __builtin_shufflevector(a_lo, a_hi,
                                                0, 1, 2, 3, 4, 5, 6, 7,
                                                8, 9, 10, 11, 12, 13, 14, 15);
            const v8h b_lo = bB[f * 64 + lane];
            const v8h b_hi = bB[f * 64 + 32 + lane];
            bf[t][ks] = __builtin_shufflevector(b_lo, b_hi,
                                                0, 1, 2, 3, 4, 5, 6, 7,
                                                8, 9, 10, 11, 12, 13, 14, 15);
        }

    // out[32x32] += A(32x64) * B(64x32): 2x2 tiles, 2 K-steps -> 8 WMMAs
    v8f acc[2][2] = {};
    #pragma unroll
    for (int ks = 0; ks < 2; ++ks)
        #pragma unroll
        for (int tm = 0; tm < 2; ++tm)
            #pragma unroll
            for (int tn = 0; tn < 2; ++tn)
                acc[tm][tn] = __builtin_amdgcn_wmma_f32_16x16x32_f16(
                    /*neg_a=*/false, af[tm][ks],
                    /*neg_b=*/false, bf[tn][ks],
                    /*c_mod=*/(short)0, acc[tm][tn],
                    /*reuse_a=*/false, /*reuse_b=*/false);

    // C/D f32 16x16 layout: lane(lo,hi), VGPR v -> (M = v + 8*hi, N = lo)
    const int lo = lane & 15;
    const int hi = lane >> 4;
    #pragma unroll
    for (int tm = 0; tm < 2; ++tm)
        #pragma unroll
        for (int tn = 0; tn < 2; ++tn)
            #pragma unroll
            for (int v = 0; v < 8; ++v)
                o_t[wave][(tm * 16 + 8 * hi + v) * 32 + tn * 16 + lo] = acc[tm][tn][v];

    __syncthreads(); // uniform: order LDS staging before copy-out

    // 4 KB per token, fully coalesced 512 B chunks, non-temporal (streaming)
    float*       op = out + (size_t)tok * DM;
    const float* sp = &o_t[wave][0];
    #pragma unroll
    for (int i = 0; i < 8; ++i) {
        const v4f val = *(const v4f*)(sp + i * 128 + lane * 4);
        __builtin_nontemporal_store(val, (v4f*)(op + i * 128 + lane * 4));
    }
}

extern "C" void kernel_launch(void* const* d_in, const int* in_sizes, int n_in,
                              void* d_out, int out_size, void* d_ws, size_t ws_size,
                              hipStream_t stream) {
    (void)in_sizes; (void)n_in; (void)out_size; (void)ws_size;

    const int*         ids   = (const int*)d_in[0];
    const signed char* c1q   = (const signed char*)d_in[1];
    const float*       s1    = (const float*)d_in[2];
    const float*       z1    = (const float*)d_in[3];
    const signed char* c2q   = (const signed char*)d_in[4];
    const float*       s2    = (const float*)d_in[5];
    const float*       z2    = (const float*)d_in[6];
    const float*       phase = (const float*)d_in[7];
    float*             outp  = (float*)d_out;
    _Float16*          ws    = (_Float16*)d_ws;  // needs (V1+V2)*2048*2 B ~= 1.8 MB

    // 1) one-time dequant + fragment swizzle of both cores (449 rows)
    tt_dequant_swizzle_kernel<<<V1 + V2, 256, 0, stream>>>(c1q, s1, z1, c2q, s2, z2,
                                                           phase, ws);
    // 2) token loop: one wave per token, exact cover -> EXEC all-ones for WMMA
    tt_embed_wmma_kernel<<<TOKENS / WAVES_PER_BLOCK, 256, 0, stream>>>(ids, ws, outp);
}
